// MSDeformAttnTransformerEncoderOnly_88630945120261
// MI455X (gfx1250) — matrix operand order repaired and checked by reading, not compiled, not run on hardware
//
#include <hip/hip_runtime.h>
#include <stdint.h>

#define DEVFN __device__ __forceinline__

typedef __attribute__((ext_vector_type(16))) __bf16 v16bf;
typedef __attribute__((ext_vector_type(8)))  float  v8f;

// ---------------- problem constants ----------------
constexpr int BATCH = 8;
constexpr int D     = 256;
constexpr int NH    = 8;
constexpr int NL    = 4;
constexpr int NP    = 4;
constexpr int DH    = 32;
constexpr int DFF   = 1024;
constexpr int LQ    = 5440;           // 64^2 + 32^2 + 16^2 + 8^2
constexpr int M     = BATCH * LQ;     // 43520 (multiple of 128)

// ---------------- helpers ----------------
DEVFN unsigned pk_bf16(float a, float b) {
  __bf16 lo = (__bf16)a, hi = (__bf16)b;
  unsigned short ulo = __builtin_bit_cast(unsigned short, lo);
  unsigned short uhi = __builtin_bit_cast(unsigned short, hi);
  return (unsigned)ulo | ((unsigned)uhi << 16);
}

struct B128x2 { uint4 lo, hi; };
DEVFN v16bf frag_from_lds(const __bf16* a, const __bf16* b) {
  B128x2 t;
  t.lo = *reinterpret_cast<const uint4*>(a);
  t.hi = *reinterpret_cast<const uint4*>(b);
  return __builtin_bit_cast(v16bf, t);
}

// ---- CDNA5 async global->LDS copy (ASYNCcnt path, no VGPR staging) ------
// VDST = LDS byte address, VADDR = 64-bit global address (VGLOBAL encoding).
DEVFN void async_copy_b128(unsigned lds_off, const void* gptr) {
  asm volatile("global_load_async_to_lds_b128 %0, %1, off"
               :: "v"(lds_off), "v"(gptr)
               : "memory");
}
#define S_WAIT_ASYNCCNT(N) asm volatile("s_wait_asynccnt " #N ::: "memory")

DEVFN unsigned lds_addr_of(const void* p) {   // low 32 bits of generic ptr = LDS offset
  return (unsigned)(size_t)p;
}

// ---------------- fp32 -> bf16 bulk convert (weights) ----------------
__global__ __launch_bounds__(256) void cvt_bf16_kernel(
    const float4* __restrict__ in, uint2* __restrict__ out, int n4)
{
  int i = blockIdx.x * 256 + threadIdx.x;
  if (i < n4) {
    float4 v = in[i];
    out[i] = make_uint2(pk_bf16(v.x, v.y), pk_bf16(v.z, v.w));
  }
}

// ---------------- input packing: (B,C,H,W) -> (B,LQ,D) f32 + bf16 --------
__global__ __launch_bounds__(256) void pack_inputs_kernel(
    const float* __restrict__ s0, const float* __restrict__ s1,
    const float* __restrict__ s2, const float* __restrict__ s3,
    const float* __restrict__ p0, const float* __restrict__ p1,
    const float* __restrict__ p2, const float* __restrict__ p3,
    const float* __restrict__ lvl, float* __restrict__ x,
    __bf16* __restrict__ xbf, float* __restrict__ pos)
{
  const int c  = threadIdx.x;
  const int m  = blockIdx.x;            // b*LQ + q
  const int b  = m / LQ;
  const int qq = m % LQ;
  int l, i, hw; const float *sp, *pp;
  if      (qq < 4096) { l = 0; i = qq;        hw = 4096; sp = s0; pp = p0; }
  else if (qq < 5120) { l = 1; i = qq - 4096; hw = 1024; sp = s1; pp = p1; }
  else if (qq < 5376) { l = 2; i = qq - 5120; hw = 256;  sp = s2; pp = p2; }
  else                { l = 3; i = qq - 5376; hw = 64;   sp = s3; pp = p3; }
  const size_t si = ((size_t)b * D + c) * hw + i;
  const float xv = sp[si];
  x  [(size_t)m * D + c] = xv;
  xbf[(size_t)m * D + c] = (__bf16)xv;
  pos[(size_t)m * D + c] = pp[si] + lvl[l * D + c];
}

// ---------------- q_bf = bf16(x + pos) ----------------
__global__ __launch_bounds__(256) void make_qbf_kernel(
    const float4* __restrict__ x, const float4* __restrict__ p,
    uint2* __restrict__ qbf, int n4)
{
  int i = blockIdx.x * 256 + threadIdx.x;
  if (i < n4) {
    float4 a = x[i], b = p[i];
    qbf[i] = make_uint2(pk_bf16(a.x + b.x, a.y + b.y), pk_bf16(a.z + b.z, a.w + b.w));
  }
}

// ---------------- bf16 WMMA GEMM: C = A(MxK,bf16) * W(KxN,bf16) + bias ---
// 256 threads (8 waves); block tile 128x128; K step 32; double-buffered LDS
// filled by global_load_async_to_lds_b128 (ASYNCcnt pipeline).
// wave grid 4(M) x 2(N); each wave: 32x64 = 2x4 wmma accumulators.
template<int NDIM, int KDIM, int RELU, int WF32, int WBF16>
__global__ __launch_bounds__(256) void gemm_bf16_wmma_kernel(
    const __bf16* __restrict__ A, const __bf16* __restrict__ Wt,
    const float* __restrict__ bias, float* __restrict__ Cf,
    __bf16* __restrict__ Cbf)
{
  __shared__ __align__(16) __bf16 As[2][128][40];   // 32 cols, pad to 40
  __shared__ __align__(16) __bf16 Bs[2][32][136];   // 128 cols, pad to 136

  const int tid  = threadIdx.x;
  const int wave = tid >> 5;
  const int lane = tid & 31;
  const int wm   = wave & 3;          // wave row (M)
  const int wn   = wave >> 2;         // wave col (N)
  const int m0   = blockIdx.y * 128;
  const int n0   = blockIdx.x * 128;
  const int lm   = lane & 15;
  const int lh   = lane >> 4;

  v8f acc[2][4] = {};

  // stage tile t into LDS buffer bsel: 4 async b128 per thread, no VGPR data
  auto stage = [&](int t, int bsel) {
    const int k0 = t * 32;
    #pragma unroll
    for (int i = 0; i < 2; ++i) {               // A: 128x32 bf16 = 512 chunks
      const int idx = tid + i * 256;            // 0..511
      const int r   = idx >> 2;                 // 0..127
      const int c8  = (idx & 3) << 3;           // 0,8,16,24
      async_copy_b128(lds_addr_of(&As[bsel][r][c8]),
                      &A[(size_t)(m0 + r) * KDIM + k0 + c8]);
    }
    #pragma unroll
    for (int i = 0; i < 2; ++i) {               // B: 32x128 bf16 = 512 chunks
      const int idx = tid + i * 256;
      const int r   = idx >> 4;                 // 0..31
      const int c8  = (idx & 15) << 3;          // 0..120
      async_copy_b128(lds_addr_of(&Bs[bsel][r][c8]),
                      &Wt[(size_t)(k0 + r) * NDIM + n0 + c8]);
    }
  };

  constexpr int T = KDIM / 32;
  stage(0, 0);
  for (int t = 0; t < T; ++t) {
    const int cur = t & 1;
    if (t + 1 < T) {
      stage(t + 1, cur ^ 1);        // 4 new asyncs in flight
      S_WAIT_ASYNCCNT(4);           // oldest 4 (tile t) complete
    } else {
      S_WAIT_ASYNCCNT(0);
    }
    __syncthreads();

    v16bf afrag[2], bfrag[4];
    #pragma unroll
    for (int f = 0; f < 2; ++f) {
      const int row = wm * 32 + f * 16 + lm;
      // A 16x32 bf16 layout: lanes 0-15 K=[0..8)&[16..24); lanes 16-31 +8
      afrag[f] = frag_from_lds(&As[cur][row][lh * 8], &As[cur][row][16 + lh * 8]);
    }
    #pragma unroll
    for (int g = 0; g < 4; ++g) {
      const int col = wn * 64 + g * 16;
      // B 32x16 bf16 layout: lane%16 = N col; lanes 0-15 K=[0..8)&[16..24); lanes 16-31 +8
      #pragma unroll
      for (int s = 0; s < 16; ++s)
        bfrag[g][s] = Bs[cur][(s < 8) ? (lh * 8 + s) : (16 + lh * 8 + (s - 8))][col + lm];
    }
    #pragma unroll
    for (int f = 0; f < 2; ++f)
      #pragma unroll
      for (int g = 0; g < 4; ++g)
        acc[f][g] = __builtin_amdgcn_wmma_f32_16x16x32_bf16(
            false, afrag[f], false, bfrag[g], (short)0, acc[f][g], false, false);
    __syncthreads();                // release buffer `cur` for tile t+2
  }

  // epilogue: C/D layout lane -> N = lane%16, M = r + 8*(lane/16)
  #pragma unroll
  for (int g = 0; g < 4; ++g) {
    const int col = n0 + wn * 64 + g * 16 + lm;
    const float bb = bias[col];
    #pragma unroll
    for (int f = 0; f < 2; ++f) {
      #pragma unroll
      for (int r = 0; r < 8; ++r) {
        const int row = m0 + wm * 32 + f * 16 + lh * 8 + r;
        float v = acc[f][g][r] + bb;
        if (RELU) v = fmaxf(v, 0.0f);
        if (WF32)  Cf [(size_t)row * NDIM + col] = v;
        if (WBF16) Cbf[(size_t)row * NDIM + col] = (__bf16)v;
      }
    }
  }
}

// ---------------- softmax over 32 sampling weights, one wave per row ------
__global__ __launch_bounds__(256) void softmax32_kernel(float* __restrict__ attn, int rows)
{
  const int wave = blockIdx.x * 8 + (threadIdx.x >> 5);
  const int lane = threadIdx.x & 31;
  if (wave >= rows) return;
  float v = attn[(size_t)wave * 32 + lane];
  float mx = v;
  #pragma unroll
  for (int o = 16; o > 0; o >>= 1) mx = fmaxf(mx, __shfl_xor(mx, o, 32));
  const float e = __expf(v - mx);
  float s = e;
  #pragma unroll
  for (int o = 16; o > 0; o >>= 1) s += __shfl_xor(s, o, 32);
  attn[(size_t)wave * 32 + lane] = e / s;
}

// ---------------- deformable sampling: one wave per (b,q,h), lane = d -----
// accumulates fp32, stores bf16 (feeds the Wout WMMA GEMM)
__global__ __launch_bounds__(256) void deform_sample_kernel(
    const float* __restrict__ value, const float* __restrict__ off,
    const float* __restrict__ attn, __bf16* __restrict__ accbf)
{
  const int wave = blockIdx.x * 8 + (threadIdx.x >> 5);
  const int lane = threadIdx.x & 31;          // channel d
  const int h  = wave & 7;
  const int mq = wave >> 3;                   // b*LQ + q
  const int qq = mq % LQ;
  const int b  = mq / LQ;

  int i0, side;
  if      (qq < 4096) { i0 = qq;        side = 64; }
  else if (qq < 5120) { i0 = qq - 4096; side = 32; }
  else if (qq < 5376) { i0 = qq - 5120; side = 16; }
  else                { i0 = qq - 5376; side = 8;  }
  const float rx = ((i0 % side) + 0.5f) / (float)side;
  const float ry = ((i0 / side) + 0.5f) / (float)side;

  const size_t obase = ((size_t)mq * NH + h) * (NL * NP * 2);
  const size_t abase = ((size_t)mq * NH + h) * (NL * NP);
  float a = 0.0f;

  int startl = 0;
  #pragma unroll
  for (int l = 0; l < NL; ++l) {
    const int   S  = 64 >> l;                 // 64,32,16,8
    const float fS = (float)S;
    const size_t vbase = ((size_t)b * LQ + startl) * D + h * DH + lane;
    #pragma unroll
    for (int p = 0; p < NP; ++p) {
      const float ox = off[obase + (l * NP + p) * 2 + 0];
      const float oy = off[obase + (l * NP + p) * 2 + 1];
      const float w  = attn[abase + l * NP + p];
      const float sx = (rx + ox / fS) * fS - 0.5f;
      const float sy = (ry + oy / fS) * fS - 0.5f;
      const float x0 = floorf(sx), y0 = floorf(sy);
      const float dx = sx - x0,  dy = sy - y0;
      const int ix0 = (int)x0, iy0 = (int)y0;
      const int ix1 = ix0 + 1, iy1 = iy0 + 1;
      auto corner = [&](int xi, int yi, float cw) {
        const bool valid = (xi >= 0) & (xi < S) & (yi >= 0) & (yi < S);
        const int cx = min(max(xi, 0), S - 1);
        const int cy = min(max(yi, 0), S - 1);
        const float vv = value[vbase + (size_t)(cy * S + cx) * D];
        a += (valid ? cw : 0.0f) * w * vv;
      };
      corner(ix0, iy0, (1.0f - dx) * (1.0f - dy));
      corner(ix1, iy0, dx * (1.0f - dy));
      corner(ix0, iy1, (1.0f - dx) * dy);
      corner(ix1, iy1, dx * dy);
    }
    startl += S * S;
  }
  accbf[(size_t)mq * D + h * DH + lane] = (__bf16)a;
}

// ---------------- fused residual + LayerNorm, writes f32 + bf16 shadow ----
__global__ __launch_bounds__(256) void add_ln_kernel(
    float* __restrict__ x, __bf16* __restrict__ xbf,
    const float* __restrict__ res,
    const float* __restrict__ g, const float* __restrict__ bet)
{
  __shared__ float red[256];
  const int t = threadIdx.x;
  const size_t base = (size_t)blockIdx.x * D;
  const float v = x[base + t] + res[base + t];
  red[t] = v;
  __syncthreads();
  for (int s = 128; s > 0; s >>= 1) { if (t < s) red[t] += red[t + s]; __syncthreads(); }
  const float mu = red[0] * (1.0f / D);
  __syncthreads();
  const float dv = v - mu;
  red[t] = dv * dv;
  __syncthreads();
  for (int s = 128; s > 0; s >>= 1) { if (t < s) red[t] += red[t + s]; __syncthreads(); }
  const float var = red[0] * (1.0f / D);
  const float y = dv * rsqrtf(var + 1e-5f) * g[t] + bet[t];
  x  [base + t] = y;
  xbf[base + t] = (__bf16)y;
}

// ---------------- orchestration ----------------
extern "C" void kernel_launch(void* const* d_in, const int* in_sizes, int n_in,
                              void* d_out, int out_size, void* d_ws, size_t ws_size,
                              hipStream_t stream)
{
  // setup_inputs() dict order: src0,pos0,src1,pos1,src2,pos2,src3,pos3,level_enc,...
  const float* s0   = (const float*)d_in[0];
  const float* p0   = (const float*)d_in[1];
  const float* s1   = (const float*)d_in[2];
  const float* p1   = (const float*)d_in[3];
  const float* s2   = (const float*)d_in[4];
  const float* p2   = (const float*)d_in[5];
  const float* s3   = (const float*)d_in[6];
  const float* p3   = (const float*)d_in[7];
  const float* lvl  = (const float*)d_in[8];
  const float* Wv   = (const float*)d_in[9];
  const float* bv   = (const float*)d_in[10];
  const float* Wo   = (const float*)d_in[11];
  const float* bo   = (const float*)d_in[12];
  const float* Wa   = (const float*)d_in[13];
  const float* ba   = (const float*)d_in[14];
  const float* Wout = (const float*)d_in[15];
  const float* bout = (const float*)d_in[16];
  const float* ln1g = (const float*)d_in[17];
  const float* ln1b = (const float*)d_in[18];
  const float* W1   = (const float*)d_in[19];
  const float* b1   = (const float*)d_in[20];
  const float* W2   = (const float*)d_in[21];
  const float* b2   = (const float*)d_in[22];
  const float* ln2g = (const float*)d_in[23];
  const float* ln2b = (const float*)d_in[24];

  float* ws = (float*)d_ws;
  const size_t S = (size_t)M * D;               // 11,141,120 floats
  float*  x     = ws;                           // [0, S)
  float*  pos   = ws + S;                       // [S, 2S)
  float*  ext   = ws + 2 * S;                   // [2S, 3S)   FFN2 f32 out
  float*  value = ws + 3 * S;                   // [3S, 4S)
  float*  offb  = ws + 4 * S;                   // [4S, 5S)   (reused as Wout f32 out)
  float*  attn  = ws + 5 * S;                   // [5S, 5.5S)
  __bf16* h1bf  = (__bf16*)(ws + 3 * S);        // [3S, 5S): M*DFF bf16 over dead value+offb
  __bf16* xbf   = (__bf16*)(ws + 5 * S + S / 2);        // 0.5S floats
  __bf16* qbf   = (__bf16*)(ws + 6 * S);                // 0.5S floats
  __bf16* accbf = (__bf16*)(ws + 6 * S + S / 2);        // 0.5S floats
  __bf16* wbf   = (__bf16*)(ws + 7 * S);                // bf16 weights, 4.52M halfwords

  // bf16 weight pool sub-offsets (halfwords)
  __bf16* wv_bf = wbf;                     // 6*256*256
  __bf16* wo_bf = wbf + 393216;            // 6*256*256
  __bf16* wa_bf = wbf + 786432;            // 6*256*128
  __bf16* wu_bf = wbf + 983040;            // 6*256*256
  __bf16* w1_bf = wbf + 1376256;           // 6*256*1024
  __bf16* w2_bf = wbf + 2949120;           // 6*1024*256

  // one-time (per launch) weight conversion: ~4.5M elems, L2-resident after
  auto cvt = [&](const float* src, __bf16* dst, int n) {
    cvt_bf16_kernel<<<(n / 4 + 255) / 256, 256, 0, stream>>>(
        (const float4*)src, (uint2*)dst, n / 4);
  };
  cvt(Wv,   wv_bf, 6 * 256 * 256);
  cvt(Wo,   wo_bf, 6 * 256 * 256);
  cvt(Wa,   wa_bf, 6 * 256 * 128);
  cvt(Wout, wu_bf, 6 * 256 * 256);
  cvt(W1,   w1_bf, 6 * 256 * 1024);
  cvt(W2,   w2_bf, 6 * 1024 * 256);

  pack_inputs_kernel<<<M, 256, 0, stream>>>(s0, s1, s2, s3, p0, p1, p2, p3,
                                            lvl, x, xbf, pos);

  const dim3 blk(256);
  const dim3 g256(2, M / 128);
  const dim3 g128(1, M / 128);
  const dim3 g1024(8, M / 128);
  const int rows = M * NH;                      // 348160

  for (int i = 0; i < 6; ++i) {
    make_qbf_kernel<<<(int)(S / 4 / 256), 256, 0, stream>>>(
        (const float4*)x, (const float4*)pos, (uint2*)qbf, (int)(S / 4));

    gemm_bf16_wmma_kernel<256, 256, 0, 1, 0><<<g256, blk, 0, stream>>>(
        xbf, wv_bf + (size_t)i * 256 * 256, bv + i * D, value, nullptr);
    gemm_bf16_wmma_kernel<256, 256, 0, 1, 0><<<g256, blk, 0, stream>>>(
        qbf, wo_bf + (size_t)i * 256 * 256, bo + i * 256, offb, nullptr);
    gemm_bf16_wmma_kernel<128, 256, 0, 1, 0><<<g128, blk, 0, stream>>>(
        qbf, wa_bf + (size_t)i * 256 * 128, ba + i * 128, attn, nullptr);

    softmax32_kernel<<<rows / 8, 256, 0, stream>>>(attn, rows);
    deform_sample_kernel<<<rows / 8, 256, 0, stream>>>(value, offb, attn, accbf);

    // output projection -> offb (f32), then residual+LN -> x, xbf
    gemm_bf16_wmma_kernel<256, 256, 0, 1, 0><<<g256, blk, 0, stream>>>(
        accbf, wu_bf + (size_t)i * 256 * 256, bout + i * D, offb, nullptr);
    add_ln_kernel<<<M, 256, 0, stream>>>(x, xbf, offb, ln1g + i * D, ln1b + i * D);

    // FFN: h1 kept bf16-only (halves traffic), FFN2 out -> ext (f32)
    gemm_bf16_wmma_kernel<1024, 256, 1, 0, 1><<<g1024, blk, 0, stream>>>(
        xbf, w1_bf + (size_t)i * 256 * 1024, b1 + i * DFF, nullptr, h1bf);
    gemm_bf16_wmma_kernel<256, 1024, 0, 1, 0><<<g256, blk, 0, stream>>>(
        h1bf, w2_bf + (size_t)i * 1024 * 256, b2 + i * D, ext, nullptr);
    add_ln_kernel<<<M, 256, 0, stream>>>(x, xbf, ext, ln2g + i * D, ln2b + i * D);
  }

  hipMemcpyAsync(d_out, x, S * sizeof(float), hipMemcpyDeviceToDevice, stream);
}
